// SCRFDTFLiteOptModel_85186381349501
// MI455X (gfx1250) — matrix operand order repaired and checked by reading, not compiled, not run on hardware
//
#include <hip/hip_runtime.h>
#include <hip/hip_bf16.h>

typedef unsigned short u16;
typedef unsigned int   u32;
typedef __bf16 bf16;
typedef bf16  v16bf __attribute__((ext_vector_type(16)));
typedef float v8f   __attribute__((ext_vector_type(8)));
typedef u32   u32x4 __attribute__((ext_vector_type(4)));

#define INF_F __builtin_inff()

// ---------------------------------------------------------------- constants
#define PTOT      134400L          // 16*(6400+1600+400) spatial positions
#define NANCH     16800            // anchors per batch
#define BATCH     16
#define TOPK_N    1000
#define NOBJ_N    100
#define KP_THR    0.45f
#define NMS_THR   0.45f

// f32 -> bf16 round-to-nearest-even
__device__ __forceinline__ u16 f2bf(float f) {
    u32 u = __builtin_bit_cast(u32, f);
    u32 r = u + 0x7FFFu + ((u >> 16) & 1u);
    return (u16)(r >> 16);
}

struct ABfrag { union { v16bf v; u32x4 q[2]; }; };

// gfx1250 async copy: global -> LDS, 16 bytes per lane, tracked by ASYNCcnt.
__device__ __forceinline__ void async_load_b128(u32 lds_byte_addr, const void* gaddr) {
    asm volatile("global_load_async_to_lds_b128 %0, %1, off"
                 :: "v"(lds_byte_addr), "v"(gaddr)
                 : "memory");
}
__device__ __forceinline__ void wait_asynccnt0() {
    asm volatile("s_wait_asynccnt 0" ::: "memory");
}

// ---------------------------------------------------------------- prep
__global__ void cvt_bf16_kernel(const float* __restrict__ in, u16* __restrict__ out, long n) {
    long i = (long)blockIdx.x * blockDim.x + threadIdx.x;
    long stride = (long)gridDim.x * blockDim.x;
    for (; i < n; i += stride) out[i] = f2bf(in[i]);
}

// pack [rows][Nout] f32 -> [rows][Npad] bf16 (zero padded)
__global__ void pack_pad_bf16(const float* __restrict__ in, u16* __restrict__ out,
                              int rows, int Nout, int Npad) {
    long total = (long)rows * Npad;
    long i = (long)blockIdx.x * blockDim.x + threadIdx.x;
    if (i >= total) return;
    int r = (int)(i / Npad), c = (int)(i % Npad);
    out[i] = (c < Nout) ? f2bf(in[(long)r * Nout + c]) : (u16)0;
}

// ---------------------------------------------------------------- 3x3 conv, dual head, bf16 WMMA
// Block = 512 threads = 16 waves; one 16-position M-tile, wave w owns output
// channels [16w,16w+16). Activation tile (with halo zeros) staged per-ky-row
// into LDS via async global->LDS copies; all waves consume from LDS.
__global__ __launch_bounds__(512) void conv3x3_dual_wmma(
    const u16* __restrict__ xbf,   // [P][256] bf16 (level base)
    const u16* __restrict__ wp,    // [2304][256] bf16
    const u16* __restrict__ wk,    // [2304][256] bf16
    const float* __restrict__ bp,  // [256]
    const float* __restrict__ bk,  // [256]
    u16* __restrict__ pf,          // [P][256] bf16 out
    u16* __restrict__ kf,
    int H, int W, int P) {
    __shared__ u16 smem[3 * 16 * 256];            // 24 KB: 3 kx taps x 16 pos x 256 ch
    const int tid  = threadIdx.x;
    const int lane = tid & 31;
    const int wave = tid >> 5;                    // 0..15 -> n-tile
    const int obase = wave * 16;
    const int hh = lane >> 4;                     // K-half select
    const int m  = lane & 15;                     // M index (A) / N index (B,C,D)
    const long mbase = (long)blockIdx.x * 16;
    const int HW = H * W;
    const u32 smem_base = (u32)(size_t)(&smem[0]);

    v8f accP = {0,0,0,0,0,0,0,0};
    v8f accK = {0,0,0,0,0,0,0,0};
    const u32x4 zero = {0,0,0,0};

    for (int ky = 0; ky < 3; ++ky) {
        __syncthreads();   // previous-iteration readers done before overwrite
        // ---- stage: 3*16*32 transfers of 16B (8 channels) each
        for (int T = tid; T < 3 * 16 * 32; T += 512) {
            const int kx = T / (16 * 32);
            const int r  = T % (16 * 32);
            const int mm = r >> 5;
            const int c  = (r & 31) << 3;          // channel chunk start
            const long pp = mbase + mm;
            bool ok = false; long grow = 0;
            if (pp < P) {
                const int bb = (int)(pp / HW);
                const int rem = (int)(pp % HW);
                const int yy = rem / W + ky - 1;
                const int xx = rem % W + kx - 1;
                if ((unsigned)yy < (unsigned)H && (unsigned)xx < (unsigned)W) {
                    ok = true;
                    grow = ((long)bb * HW + (long)yy * W + xx) << 8;
                }
            }
            const int sidx = ((kx * 16 + mm) << 8) + c;
            if (ok)  async_load_b128(smem_base + (u32)(sidx * 2), (const void*)(xbf + grow + c));
            else     *(u32x4*)&smem[sidx] = zero;  // halo / tail zeros
        }
        wait_asynccnt0();
        __syncthreads();
        // ---- consume: 3 kx taps x 8 k-blocks, branch-free
        for (int kx = 0; kx < 3; ++kx) {
            const int tap = ky * 3 + kx;
            const u16* arow = &smem[((kx * 16) + m) << 8];
            const long krow = ((long)tap * 256 + lane) << 8;
            __builtin_prefetch(wp + krow + 8192, 0, 1);
#pragma unroll
            for (int kb = 0; kb < 8; ++kb) {
                ABfrag a, bP, bK;
                a.q[0] = *(const u32x4*)(arow + kb * 32 + 8 * hh);
                a.q[1] = *(const u32x4*)(arow + kb * 32 + 16 + 8 * hh);
                const long koff = krow + ((long)kb << 13);   // kb*32 rows * 256
                const u16* wpP = wp + koff - ((long)lane << 8) + obase + m;
                const u16* wkP = wk + koff - ((long)lane << 8) + obase + m;
#pragma unroll
                for (int s = 0; s < 16; ++s) {
                    const int kk = (s < 8) ? (8 * hh + s) : (16 + 8 * hh + s - 8);
                    bP.v[s] = __builtin_bit_cast(bf16, wpP[(long)kk << 8]);
                    bK.v[s] = __builtin_bit_cast(bf16, wkP[(long)kk << 8]);
                }
                accP = __builtin_amdgcn_wmma_f32_16x16x32_bf16(false, a.v, false, bP.v,
                                                               (short)0, accP, false, false);
                accK = __builtin_amdgcn_wmma_f32_16x16x32_bf16(false, a.v, false, bK.v,
                                                               (short)0, accK, false, false);
            }
        }
    }
    const float biasP = bp[obase + m];
    const float biasK = bk[obase + m];
#pragma unroll
    for (int v = 0; v < 8; ++v) {
        const long pr = mbase + v + 8 * hh;
        if (pr < P) {
            float fp = accP[v] + biasP; fp = fp > 0.f ? fp : 0.f;
            float fk = accK[v] + biasK; fk = fk > 0.f ? fk : 0.f;
            pf[(pr << 8) + obase + m] = f2bf(fp);
            kf[(pr << 8) + obase + m] = f2bf(fk);
        }
    }
}

// ---------------------------------------------------------------- 1x1 head, bf16 WMMA, K=256
__global__ __launch_bounds__(32) void head1x1_wmma(
    const u16* __restrict__ act,   // [P][256] bf16
    const u16* __restrict__ wgt,   // [256][Npad] bf16
    const float* __restrict__ bias,// [Nout]
    const float* __restrict__ sptr, float smul, int level,
    float* __restrict__ out,       // [P][Npad]
    int P, int Npad, int Nout) {
    const int lane  = threadIdx.x & 31;
    const int obase = blockIdx.y * 16;
    const int hh = lane >> 4, m = lane & 15;
    const long mbase = (long)blockIdx.x * 16;
    const long pos = mbase + m;
    const bool pvalid = pos < P;
    const u16* arow = act + (pos << 8);
    v8f acc = {0,0,0,0,0,0,0,0};
    const u32x4 zero = {0,0,0,0};
#pragma unroll
    for (int kb = 0; kb < 8; ++kb) {
        ABfrag a, bF;
        if (pvalid) {
            const u16* ap = arow + kb * 32 + 8 * hh;
            a.q[0] = *(const u32x4*)(ap);
            a.q[1] = *(const u32x4*)(ap + 16);
        } else { a.q[0] = zero; a.q[1] = zero; }
        const u16* wP = wgt + (long)(kb * 32) * Npad + obase + m;
#pragma unroll
        for (int s = 0; s < 16; ++s) {
            const int kk = (s < 8) ? (8 * hh + s) : (16 + 8 * hh + s - 8);
            bF.v[s] = __builtin_bit_cast(bf16, wP[(long)kk * Npad]);
        }
        acc = __builtin_amdgcn_wmma_f32_16x16x32_bf16(false, a.v, false, bF.v,
                                                      (short)0, acc, false, false);
    }
    const float scale = smul * (sptr ? sptr[level] : 1.0f);
    const float bia = (obase + m < Nout) ? bias[obase + m] : 0.f;
#pragma unroll
    for (int v = 0; v < 8; ++v) {
        const long pr = mbase + v + 8 * hh;
        if (pr < P) out[pr * Npad + obase + m] = (acc[v] + bia) * scale;
    }
}

// ---------------------------------------------------------------- anchor decode helper
__device__ __forceinline__ void decode_anchor(int a, int b, long& gp, int& anchor,
                                              float& cx, float& cy) {
    if (a < 12800) {
        int idx = a, cell = idx >> 1; anchor = idx & 1;
        gp = 0L + (long)b * 6400 + cell;
        cx = (float)(cell % 80) * 8.f;  cy = (float)(cell / 80) * 8.f;
    } else if (a < 16000) {
        int idx = a - 12800, cell = idx >> 1; anchor = idx & 1;
        gp = 102400L + (long)b * 1600 + cell;
        cx = (float)(cell % 40) * 16.f; cy = (float)(cell / 40) * 16.f;
    } else {
        int idx = a - 16000, cell = idx >> 1; anchor = idx & 1;
        gp = 128000L + (long)b * 400 + cell;
        cx = (float)(cell % 20) * 32.f; cy = (float)(cell / 20) * 32.f;
    }
}

// ---------------------------------------------------------------- score concat
__global__ void build_scores(const float* __restrict__ c0, const float* __restrict__ c1,
                             const float* __restrict__ c2, float* __restrict__ scores) {
    long i = (long)blockIdx.x * blockDim.x + threadIdx.x;
    if (i >= (long)BATCH * NANCH) return;
    int b = (int)(i / NANCH), a = (int)(i % NANCH);
    float s;
    if (a < 12800)      s = c0[(long)b * 12800 + a];
    else if (a < 16000) s = c1[(long)b * 3200 + (a - 12800)];
    else                s = c2[(long)b * 800  + (a - 16000)];
    scores[i] = s;
}

// ---------------------------------------------------------------- top-1000 (stable selection)
__global__ __launch_bounds__(1024) void topk_kernel(const float* __restrict__ scores,
                                                    float* __restrict__ work,
                                                    float* __restrict__ top_sc,
                                                    int* __restrict__ top_idx) {
    const int b = blockIdx.x, tid = threadIdx.x;
    for (int i = tid; i < NANCH; i += 1024) {
        float s = scores[(long)b * NANCH + i];
        work[(long)b * NANCH + i] = (s > KP_THR) ? s : -INF_F;
    }
    __syncthreads();
    __shared__ float sval[1024];
    __shared__ int   sidx[1024];
    for (int t = 0; t < TOPK_N; ++t) {
        float best = -INF_F; int bi = NANCH;
        for (int i = tid; i < NANCH; i += 1024) {
            float v = work[(long)b * NANCH + i];
            if (v > best || (v == best && i < bi)) { best = v; bi = i; }
        }
        sval[tid] = best; sidx[tid] = bi;
        __syncthreads();
        for (int s = 512; s > 0; s >>= 1) {
            if (tid < s) {
                float v2 = sval[tid + s]; int i2 = sidx[tid + s];
                if (v2 > sval[tid] || (v2 == sval[tid] && i2 < sidx[tid])) {
                    sval[tid] = v2; sidx[tid] = i2;
                }
            }
            __syncthreads();
        }
        if (tid == 0) {
            int wi = (sidx[0] >= NANCH) ? 0 : sidx[0];
            top_sc[b * TOPK_N + t]  = sval[0];
            top_idx[b * TOPK_N + t] = wi;
            if (sidx[0] < NANCH) work[(long)b * NANCH + wi] = -INF_F;
        }
        __syncthreads();
    }
}

// ---------------------------------------------------------------- gather + box/kps decode
__global__ void gather_decode(const int* __restrict__ top_idx,
                              const float* __restrict__ bbo,   // [P][16]
                              const float* __restrict__ kpo,   // [P][32]
                              const float* __restrict__ origin,// [16][2]
                              float* __restrict__ tb,          // [16][1000][4]
                              float* __restrict__ tk) {        // [16][1000][10]
    long i = (long)blockIdx.x * blockDim.x + threadIdx.x;
    if (i >= (long)BATCH * TOPK_N) return;
    int b = (int)(i / TOPK_N);
    int a = top_idx[i];
    long gp; int anchor; float cx, cy;
    decode_anchor(a, b, gp, anchor, cx, cy);
    const float* d = bbo + gp * 16 + anchor * 4;   // already * s_bbox * stride
    float x1 = cx - d[0], y1 = cy - d[1], x2 = cx + d[2], y2 = cy + d[3];
    float rh = origin[b * 2 + 0] * (1.f / 640.f);
    float rw = origin[b * 2 + 1] * (1.f / 640.f);
    tb[i * 4 + 0] = y1 * rh; tb[i * 4 + 1] = x1 * rw;
    tb[i * 4 + 2] = y2 * rh; tb[i * 4 + 3] = x2 * rw;
    const float* k = kpo + gp * 32 + anchor * 10;  // already * stride
#pragma unroll
    for (int u = 0; u < 5; ++u) {
        tk[i * 10 + 2 * u + 0] = cx + k[2 * u + 0];
        tk[i * 10 + 2 * u + 1] = cy + k[2 * u + 1];
    }
}

// ---------------------------------------------------------------- 3DMM decode: one block per (b,t)
__global__ __launch_bounds__(64) void tddm_kernel(const int* __restrict__ top_idx,
                                                  const float* __restrict__ par, // [P][128]
                                                  const float* __restrict__ pms, // [2][59]
                                                  const float* __restrict__ u_base,
                                                  const float* __restrict__ shp, // [204][40]
                                                  const float* __restrict__ expb,// [204][10]
                                                  float* __restrict__ lnm) {     // [16][1000][136]
    const int i = blockIdx.x, tid = threadIdx.x;
    const int b = i / TOPK_N;
    const int a = top_idx[i];
    long gp; int anchor; float cx, cy;
    decode_anchor(a, b, gp, anchor, cx, cy);
    __shared__ float p[59];
    __shared__ float verts[204];
    if (tid < 59)
        p[tid] = par[gp * 128 + anchor * 59 + tid] * pms[59 + tid] + pms[tid];
    __syncthreads();
    for (int v = tid; v < 204; v += 64) {
        float s = u_base[v];
        const float* sr = shp + (long)v * 40;
#pragma unroll 8
        for (int j = 0; j < 40; ++j) s += sr[j] * p[9 + j];
        const float* er = expb + (long)v * 10;
#pragma unroll
        for (int j = 0; j < 10; ++j) s += er[j] * p[49 + j];
        verts[v] = s;
    }
    __syncthreads();
    for (int o = tid; o < 136; o += 64) {
        int v = o >> 1, d = o & 1;
        float s = verts[v * 3 + 0] * p[d * 3 + 0]
                + verts[v * 3 + 1] * p[d * 3 + 1]
                + verts[v * 3 + 2] * p[d * 3 + 2];
        lnm[(long)i * 136 + o] = s;
    }
}

// ---------------------------------------------------------------- NMS (serial i, parallel j)
__global__ __launch_bounds__(1024) void nms_kernel(const float* __restrict__ tb,
                                                   const float* __restrict__ top_sc,
                                                   int* __restrict__ keep) {
    const int b = blockIdx.x, tid = threadIdx.x;
    __shared__ float bx[TOPK_N * 4];
    __shared__ float area[TOPK_N];
    __shared__ int   kp[TOPK_N];
    __shared__ int   sup;
    for (int i = tid; i < TOPK_N; i += 1024) {
        float b0 = tb[((long)b * TOPK_N + i) * 4 + 0];
        float b1 = tb[((long)b * TOPK_N + i) * 4 + 1];
        float b2 = tb[((long)b * TOPK_N + i) * 4 + 2];
        float b3 = tb[((long)b * TOPK_N + i) * 4 + 3];
        bx[i * 4 + 0] = b0; bx[i * 4 + 1] = b1; bx[i * 4 + 2] = b2; bx[i * 4 + 3] = b3;
        area[i] = fmaxf(b2 - b0, 0.f) * fmaxf(b3 - b1, 0.f);
        kp[i] = 0;
    }
    __syncthreads();
    for (int i = 0; i < TOPK_N; ++i) {
        if (tid == 0) sup = 0;
        __syncthreads();
        for (int j = tid; j < i; j += 1024) {
            if (kp[j]) {
                float y1 = fmaxf(bx[i * 4 + 0], bx[j * 4 + 0]);
                float x1 = fmaxf(bx[i * 4 + 1], bx[j * 4 + 1]);
                float y2 = fminf(bx[i * 4 + 2], bx[j * 4 + 2]);
                float x2 = fminf(bx[i * 4 + 3], bx[j * 4 + 3]);
                float inter = fmaxf(y2 - y1, 0.f) * fmaxf(x2 - x1, 0.f);
                float iou = inter / (area[i] + area[j] - inter + 1e-9f);
                if (iou > NMS_THR) atomicOr(&sup, 1);
            }
        }
        __syncthreads();
        if (tid == 0) {
            bool valid = top_sc[b * TOPK_N + i] != -INF_F;
            kp[i] = (valid && !sup) ? 1 : 0;
        }
        __syncthreads();
    }
    for (int i = tid; i < TOPK_N; i += 1024) keep[b * TOPK_N + i] = kp[i];
}

// ---------------------------------------------------------------- final top-100 + output write
__global__ __launch_bounds__(256) void final_kernel(const float* __restrict__ top_sc,
                                                    const int* __restrict__ keep,
                                                    const float* __restrict__ tb,
                                                    const float* __restrict__ tk,
                                                    const float* __restrict__ lnm,
                                                    float* __restrict__ out) {
    const int b = blockIdx.x, tid = threadIdx.x;
    __shared__ float w[TOPK_N];
    __shared__ float sval[256];
    __shared__ int   sidx[256];
    __shared__ int   selIdx[NOBJ_N];
    __shared__ float selSc[NOBJ_N];
    for (int i = tid; i < TOPK_N; i += 256)
        w[i] = keep[b * TOPK_N + i] ? top_sc[b * TOPK_N + i] : -INF_F;
    __syncthreads();
    for (int t = 0; t < NOBJ_N; ++t) {
        float best = -INF_F; int bi = TOPK_N;
        for (int i = tid; i < TOPK_N; i += 256) {
            float v = w[i];
            if (v > best || (v == best && i < bi)) { best = v; bi = i; }
        }
        sval[tid] = best; sidx[tid] = bi;
        __syncthreads();
        for (int s = 128; s > 0; s >>= 1) {
            if (tid < s) {
                float v2 = sval[tid + s]; int i2 = sidx[tid + s];
                if (v2 > sval[tid] || (v2 == sval[tid] && i2 < sidx[tid])) {
                    sval[tid] = v2; sidx[tid] = i2;
                }
            }
            __syncthreads();
        }
        if (tid == 0) {
            int wi = (sidx[0] >= TOPK_N) ? 0 : sidx[0];
            selIdx[t] = wi; selSc[t] = sval[0];
            if (sidx[0] < TOPK_N) w[wi] = -INF_F;
        }
        __syncthreads();
    }
    const long LNM_OFF = (long)BATCH * NOBJ_N * 5;              // 8000
    const long KPS_OFF = LNM_OFF + (long)BATCH * NOBJ_N * 136;  // 225600
    for (int e = tid; e < NOBJ_N * 5; e += 256) {
        int t = e / 5, j = e % 5;
        bool valid = selSc[t] != -INF_F;
        float val = -1.f;
        if (valid) val = (j < 4) ? tb[((long)b * TOPK_N + selIdx[t]) * 4 + j] : selSc[t];
        out[(long)(b * NOBJ_N + t) * 5 + j] = val;
    }
    for (int e = tid; e < NOBJ_N * 136; e += 256) {
        int t = e / 136, j = e % 136;
        bool valid = selSc[t] != -INF_F;
        out[LNM_OFF + (long)(b * NOBJ_N + t) * 136 + j] =
            valid ? lnm[((long)b * TOPK_N + selIdx[t]) * 136 + j] : INF_F;
    }
    for (int e = tid; e < NOBJ_N * 10; e += 256) {
        int t = e / 10, j = e % 10;
        bool valid = selSc[t] != -INF_F;
        out[KPS_OFF + (long)(b * NOBJ_N + t) * 10 + j] =
            valid ? tk[((long)b * TOPK_N + selIdx[t]) * 10 + j] : -1.f;
    }
}

// ================================================================ launch
static inline size_t alignup(size_t v) { return (v + 255) & ~(size_t)255; }

extern "C" void kernel_launch(void* const* d_in, const int* in_sizes, int n_in,
                              void* d_out, int out_size, void* d_ws, size_t ws_size,
                              hipStream_t stream) {
    (void)in_sizes; (void)n_in; (void)out_size; (void)ws_size;
    const float* cls0 = (const float*)d_in[0];
    const float* reg0 = (const float*)d_in[1];
    const float* x0   = (const float*)d_in[2];
    const float* cls1 = (const float*)d_in[3];
    const float* reg1 = (const float*)d_in[4];
    const float* x1   = (const float*)d_in[5];
    const float* cls2 = (const float*)d_in[6];
    const float* reg2 = (const float*)d_in[7];
    const float* x2   = (const float*)d_in[8];
    const float* origin = (const float*)d_in[9];
    const float* w_bbox = (const float*)d_in[10];
    const float* b_bbox = (const float*)d_in[11];
    const float* s_bbox = (const float*)d_in[12];
    const float* w_p3 = (const float*)d_in[13];
    const float* b_p3 = (const float*)d_in[14];
    const float* w_pp = (const float*)d_in[15];
    const float* b_pp = (const float*)d_in[16];
    const float* w_k3 = (const float*)d_in[17];
    const float* b_k3 = (const float*)d_in[18];
    const float* w_kp = (const float*)d_in[19];
    const float* b_kp = (const float*)d_in[20];
    const float* pms  = (const float*)d_in[21];
    const float* u_base = (const float*)d_in[22];
    const float* shp  = (const float*)d_in[23];
    const float* expb = (const float*)d_in[24];
    float* out = (float*)d_out;

    // ---- workspace carve-up
    char* ws = (char*)d_ws;
    size_t off = 0;
    auto take = [&](size_t bytes) { char* p = ws + off; off += alignup(bytes); return p; };
    u16* xbf   = (u16*)take(PTOT * 256 * 2);
    u16* regbf = (u16*)take(PTOT * 256 * 2);
    u16* wp3bf = (u16*)take(3L * 2304 * 256 * 2);
    u16* wk3bf = (u16*)take(3L * 2304 * 256 * 2);
    u16* wppbf = (u16*)take(3L * 256 * 128 * 2);
    u16* wkpbf = (u16*)take(3L * 256 * 32 * 2);
    u16* wbbbf = (u16*)take(3L * 256 * 16 * 2);
    u16* pfbf  = (u16*)take(PTOT * 256 * 2);
    u16* kfbf  = (u16*)take(PTOT * 256 * 2);
    float* par = (float*)take(PTOT * 128 * 4);
    float* kpo = (float*)take(PTOT * 32 * 4);
    float* bbo = (float*)take(PTOT * 16 * 4);
    float* scores = (float*)take((long)BATCH * NANCH * 4);
    float* work   = (float*)take((long)BATCH * NANCH * 4);
    float* top_sc = (float*)take((long)BATCH * TOPK_N * 4);
    int*   top_idx= (int*)  take((long)BATCH * TOPK_N * 4);
    float* tb     = (float*)take((long)BATCH * TOPK_N * 4 * 4);
    float* tk     = (float*)take((long)BATCH * TOPK_N * 10 * 4);
    float* lnm    = (float*)take((long)BATCH * TOPK_N * 136 * 4);
    int*   keep   = (int*)  take((long)BATCH * TOPK_N * 4);

    const long Poff[3]  = {0, 102400, 128000};
    const long Plev[3]  = {102400, 25600, 6400};
    const int  Hlev[3]  = {80, 40, 20};
    const int  Wlev[3]  = {80, 40, 20};
    const float strd[3] = {8.f, 16.f, 32.f};
    const float* xin[3] = {x0, x1, x2};
    const float* rin[3] = {reg0, reg1, reg2};

    // ---- bf16 conversions
    auto cvt = [&](const float* in, u16* outp, long n) {
        int blocks = (int)((n + 255) / 256);
        hipLaunchKernelGGL(cvt_bf16_kernel, dim3(blocks), dim3(256), 0, stream, in, outp, n);
    };
    for (int i = 0; i < 3; ++i) {
        cvt(xin[i], xbf + Poff[i] * 256, Plev[i] * 256);
        cvt(rin[i], regbf + Poff[i] * 256, Plev[i] * 256);
    }
    cvt(w_p3, wp3bf, 3L * 2304 * 256);
    cvt(w_k3, wk3bf, 3L * 2304 * 256);
    {
        long n;
        n = 3L * 256 * 128;
        hipLaunchKernelGGL(pack_pad_bf16, dim3((int)((n + 255) / 256)), dim3(256), 0, stream,
                           w_pp, wppbf, 3 * 256, 118, 128);
        n = 3L * 256 * 32;
        hipLaunchKernelGGL(pack_pad_bf16, dim3((int)((n + 255) / 256)), dim3(256), 0, stream,
                           w_kp, wkpbf, 3 * 256, 20, 32);
        n = 3L * 256 * 16;
        hipLaunchKernelGGL(pack_pad_bf16, dim3((int)((n + 255) / 256)), dim3(256), 0, stream,
                           w_bbox, wbbbf, 3 * 256, 8, 16);
    }

    // ---- 3x3 dual conv (WMMA + async LDS staging) per level
    for (int i = 0; i < 3; ++i) {
        int mtiles = (int)((Plev[i] + 15) / 16);
        hipLaunchKernelGGL(conv3x3_dual_wmma, dim3(mtiles), dim3(512), 0, stream,
                           xbf + Poff[i] * 256,
                           wp3bf + (long)i * 2304 * 256, wk3bf + (long)i * 2304 * 256,
                           b_p3 + i * 256, b_k3 + i * 256,
                           pfbf + Poff[i] * 256, kfbf + Poff[i] * 256,
                           Hlev[i], Wlev[i], (int)Plev[i]);
    }

    // ---- 1x1 heads (WMMA) per level
    for (int i = 0; i < 3; ++i) {
        int mtiles = (int)((Plev[i] + 15) / 16);
        hipLaunchKernelGGL(head1x1_wmma, dim3(mtiles, 8), dim3(32), 0, stream,
                           pfbf + Poff[i] * 256, wppbf + (long)i * 256 * 128,
                           b_pp + i * 118, (const float*)nullptr, 1.0f, i,
                           par + Poff[i] * 128, (int)Plev[i], 128, 118);
        hipLaunchKernelGGL(head1x1_wmma, dim3(mtiles, 2), dim3(32), 0, stream,
                           kfbf + Poff[i] * 256, wkpbf + (long)i * 256 * 32,
                           b_kp + i * 20, (const float*)nullptr, strd[i], i,
                           kpo + Poff[i] * 32, (int)Plev[i], 32, 20);
        hipLaunchKernelGGL(head1x1_wmma, dim3(mtiles, 1), dim3(32), 0, stream,
                           regbf + Poff[i] * 256, wbbbf + (long)i * 256 * 16,
                           b_bbox + i * 8, s_bbox, strd[i], i,
                           bbo + Poff[i] * 16, (int)Plev[i], 16, 8);
    }

    // ---- postprocess
    {
        long n = (long)BATCH * NANCH;
        hipLaunchKernelGGL(build_scores, dim3((int)((n + 255) / 256)), dim3(256), 0, stream,
                           cls0, cls1, cls2, scores);
    }
    hipLaunchKernelGGL(topk_kernel, dim3(BATCH), dim3(1024), 0, stream,
                       scores, work, top_sc, top_idx);
    {
        long n = (long)BATCH * TOPK_N;
        hipLaunchKernelGGL(gather_decode, dim3((int)((n + 255) / 256)), dim3(256), 0, stream,
                           top_idx, bbo, kpo, origin, tb, tk);
    }
    hipLaunchKernelGGL(tddm_kernel, dim3(BATCH * TOPK_N), dim3(64), 0, stream,
                       top_idx, par, pms, u_base, shp, expb, lnm);
    hipLaunchKernelGGL(nms_kernel, dim3(BATCH), dim3(1024), 0, stream, tb, top_sc, keep);
    hipLaunchKernelGGL(final_kernel, dim3(BATCH), dim3(256), 0, stream,
                       top_sc, keep, tb, tk, lnm, out);
}
